// SimpleMoE_55679956025635
// MI455X (gfx1250) — compile-verified, matrix operand order checked by reading, not run on hardware
//
#include <hip/hip_runtime.h>
#include <cstdint>
#include <cstddef>

// ---------------------------------------------------------------------------
// SimpleMoE for MI455X (gfx1250): dense 8-expert MoE, bf16 WMMA path.
//   out[n,o] = sum_e softmax(x@gw.T+gb)[n,e] * (x @ W_e.T + b_e)[n,o]
// Gate folded into the A operand per expert (exact in f32 accumulation):
//   out = sum_{e,k} (g_e ⊙ A_k) * B_{e,k}   (+ sum_e g_e*b_e in the epilogue)
//
// Block: 512 threads = 16 wave32, tile BM=256 x BN=64, BK=32.
// Each wave owns a 32x32 patch = 2x2 WMMA accumulators (low VGPR pressure,
// no VGPR-MSB mode, 4+ waves/SIMD for latency hiding).
// Tiles fed by the Tensor Data Mover (tensor_load_to_lds) with hardware row
// padding to 40 elements (80B = 20 dwords -> conflict-free LDS banking),
// double-buffered with s_wait_tensorcnt + workgroup barriers.
// ---------------------------------------------------------------------------

#define NTOK 8192
#define DIN  1024
#define DOUT 1024
#define NEXP 8

#define BM 256
#define BN 64
#define BK 32
#define NKI (DIN / BK)   // 32 k-iterations
#define TPAD 40          // padded LDS row length (elements); 80B = 20 dwords
#define NTHREADS 512

typedef __attribute__((ext_vector_type(16))) __bf16 v16bf;
typedef __attribute__((ext_vector_type(8)))  __bf16 v8bf;
typedef __attribute__((ext_vector_type(8)))  float  v8f;
typedef __attribute__((ext_vector_type(4)))  unsigned int u32x4;
typedef __attribute__((ext_vector_type(8)))  int    i32x8;
typedef __attribute__((ext_vector_type(4)))  int    i32x4;

#if defined(__HIP_DEVICE_COMPILE__)
#  if __has_builtin(__builtin_amdgcn_tensor_load_to_lds)
#    define USE_TDM 1
#  endif
#endif
#ifndef USE_TDM
#define USE_TDM 0
#endif

#define CAT16(lo, hi) __builtin_shufflevector((lo), (hi), \
    0, 1, 2, 3, 4, 5, 6, 7, 8, 9, 10, 11, 12, 13, 14, 15)

__device__ __forceinline__ unsigned short f32_bf16(float f) {
    unsigned int u = __float_as_uint(f);
    u += 0x7FFFu + ((u >> 16) & 1u);          // round-to-nearest-even
    return (unsigned short)(u >> 16);
}

#if USE_TDM
// Build a D# (ISA ch.8) for a rows x 32 bf16 tile, row stride DIN elements,
// destination LDS rows padded to TPAD=40 elements via pad_interval=16dw(code 3),
// pad_amount=4dw(code 3): 16dw data + 4dw pad = 20dw = 40 elements per row.
__device__ __forceinline__ void tdm_load_tile(unsigned lds_off,
                                              const unsigned short* gsrc,
                                              int rows) {
    unsigned long long ga = (unsigned long long)(uintptr_t)gsrc;
    unsigned alo = (unsigned)__builtin_amdgcn_readfirstlane((int)(unsigned)ga);
    unsigned ahi = (unsigned)__builtin_amdgcn_readfirstlane(
                       (int)((unsigned)(ga >> 32) & 0x1FFFFFFu));
    lds_off = (unsigned)__builtin_amdgcn_readfirstlane((int)lds_off);

    u32x4 g0;
    g0[0] = 1u;                    // count=1, user mode, no gather
    g0[1] = lds_off;               // lds_addr (bytes)
    g0[2] = alo;                   // global_addr[31:0]
    g0[3] = ahi | (2u << 30);      // global_addr[56:32] | type=2 (image)

    const unsigned td0 = DIN;      // tensor_dim0 (elements)
    const unsigned td1 = 1u << 20; // tensor_dim1 (generous, tiles always in-bounds)
    i32x8 g1;
    // data_size=1 (2B) | pad_enable | pad_interval code 3 (16dw) | pad_amount code 3 (4dw)
    g1[0] = (int)((1u << 16) | (1u << 20) | (3u << 22) | (3u << 25));
    g1[1] = (int)((td0 & 0xFFFFu) << 16);                 // tensor_dim0[15:0]
    g1[2] = (int)((td0 >> 16) | ((td1 & 0xFFFFu) << 16)); // dim0 hi | dim1 lo
    g1[3] = (int)((td1 >> 16) | ((unsigned)BK << 16));    // dim1 hi | tile_dim0=32
    g1[4] = rows;                                         // tile_dim1 (tile_dim2=0)
    g1[5] = (int)DIN;                                     // tensor_dim0_stride lo
    g1[6] = 0;
    g1[7] = 0;
    const i32x4 z4 = {0, 0, 0, 0};
#if __clang_major__ >= 23
    const i32x8 z8 = {0, 0, 0, 0, 0, 0, 0, 0};
    __builtin_amdgcn_tensor_load_to_lds(g0, g1, z4, z4, z8, 0);
#else
    __builtin_amdgcn_tensor_load_to_lds(g0, g1, z4, z4, 0);
#endif
}
#endif // USE_TDM

// ---------------------------------------------------------------------------
// Kernel 1: f32 -> bf16 (8 elements / thread, sizes divide exactly)
// ---------------------------------------------------------------------------
__global__ void __launch_bounds__(256)
moe_cvt_bf16(const float* __restrict__ src, unsigned short* __restrict__ dst) {
    const size_t i = ((size_t)blockIdx.x * 256u + threadIdx.x) * 8u;
    const float4 a = *(const float4*)(src + i);
    const float4 b = *(const float4*)(src + i + 4);
    union { unsigned short h[8]; uint4 v; } o;
    o.h[0] = f32_bf16(a.x); o.h[1] = f32_bf16(a.y);
    o.h[2] = f32_bf16(a.z); o.h[3] = f32_bf16(a.w);
    o.h[4] = f32_bf16(b.x); o.h[5] = f32_bf16(b.y);
    o.h[6] = f32_bf16(b.z); o.h[7] = f32_bf16(b.w);
    *(uint4*)(dst + i) = o.v;
}

// ---------------------------------------------------------------------------
// Kernel 2: gates = softmax(x @ gate_w.T + gate_b)  -- one wave32 per token
// ---------------------------------------------------------------------------
__global__ void __launch_bounds__(256)
moe_gates(const float* __restrict__ x, const float* __restrict__ gw,
          const float* __restrict__ gb, float* __restrict__ gates) {
    __shared__ float gwl[NEXP * DIN];
    for (int i = threadIdx.x; i < NEXP * DIN; i += 256) gwl[i] = gw[i];
    __syncthreads();

    const int wave = threadIdx.x >> 5;
    const int lane = threadIdx.x & 31;
    const int tok  = blockIdx.x * 8 + wave;
    const float* xr = x + (size_t)tok * DIN;

    float acc[NEXP];
#pragma unroll
    for (int e = 0; e < NEXP; ++e) acc[e] = 0.f;
    for (int i = lane; i < DIN; i += 32) {
        const float xv = xr[i];
#pragma unroll
        for (int e = 0; e < NEXP; ++e)
            acc[e] = __builtin_fmaf(xv, gwl[e * DIN + i], acc[e]);
    }
#pragma unroll
    for (int e = 0; e < NEXP; ++e) {
#pragma unroll
        for (int off = 16; off > 0; off >>= 1)
            acc[e] += __shfl_xor(acc[e], off, 32);
        acc[e] += gb[e];
    }
    float m = acc[0];
#pragma unroll
    for (int e = 1; e < NEXP; ++e) m = fmaxf(m, acc[e]);
    float s = 0.f;
#pragma unroll
    for (int e = 0; e < NEXP; ++e) { acc[e] = __expf(acc[e] - m); s += acc[e]; }
    const float inv = 1.0f / s;
#pragma unroll
    for (int e = 0; e < NEXP; ++e)
        if (lane == e) gates[(size_t)tok * NEXP + e] = acc[e] * inv;
}

// ---------------------------------------------------------------------------
// Kernel 3: main gated GEMM. 512 thr = 16 waves; each wave: 2x2 16x16 tiles.
// ---------------------------------------------------------------------------
__global__ void __launch_bounds__(NTHREADS)
moe_wmma(const unsigned short* __restrict__ xh,   // [NTOK][DIN] bf16
         const unsigned short* __restrict__ wh,   // [E][DOUT][DIN] bf16
         const float* __restrict__ eb,            // [E][DOUT]
         const float* __restrict__ gates,         // [NTOK][E]
         float* __restrict__ out) {               // [NTOK][DOUT]
#if defined(__HIP_DEVICE_COMPILE__)
    __shared__ __align__(16) unsigned short Xs[2][BM * TPAD];  // 40 KB
    __shared__ __align__(16) unsigned short Ws[2][BN * TPAD];  // 10 KB

    const int tid  = threadIdx.x;
    const int wave = tid >> 5;      // 0..15
    const int lane = tid & 31;
    const int lr   = lane & 15;     // row within 16-wide fragment
    const int lh   = lane >> 4;     // K-half select
    const int wm   = wave >> 1;     // 0..7  -> 32-row M group
    const int ng   = wave & 1;      // 0..1  -> 32-col N group
    const int m0   = blockIdx.x * BM;
    const int n0   = blockIdx.y * BN;

    // Gate values for A-row scaling: A-fragment lane holds row M = lr.
    __bf16 gA[2][NEXP];
#pragma unroll
    for (int mi = 0; mi < 2; ++mi) {
        const float* gp = gates + (size_t)(m0 + 32 * wm + 16 * mi + lr) * NEXP;
        const float4 glo = *(const float4*)gp;
        const float4 ghi = *(const float4*)(gp + 4);
        gA[mi][0] = (__bf16)glo.x; gA[mi][1] = (__bf16)glo.y;
        gA[mi][2] = (__bf16)glo.z; gA[mi][3] = (__bf16)glo.w;
        gA[mi][4] = (__bf16)ghi.x; gA[mi][5] = (__bf16)ghi.y;
        gA[mi][6] = (__bf16)ghi.z; gA[mi][7] = (__bf16)ghi.w;
    }

    auto issueX = [&](int buf, int kk) {
#if USE_TDM
        if (wave == 0)
            tdm_load_tile((unsigned)(uintptr_t)&Xs[buf][0],
                          xh + (size_t)m0 * DIN + kk * BK, BM);
#else
#pragma unroll
        for (int c = 0; c < 2; ++c) {
            const int idx = (c << 9) | tid;         // 1024 chunks of 8 elems
            const int row = idx >> 2;               // 0..255
            const int cc  = (idx & 3) << 3;         // 0,8,16,24
            *(uint4*)&Xs[buf][row * TPAD + cc] =
                *(const uint4*)(xh + (size_t)(m0 + row) * DIN + kk * BK + cc);
        }
#endif
    };
    auto issueW = [&](int buf, int e, int kk) {
#if USE_TDM
        if (wave == 0)
            tdm_load_tile((unsigned)(uintptr_t)&Ws[buf][0],
                          wh + ((size_t)e * DOUT + n0) * DIN + kk * BK, BN);
#else
        if (tid < 256) {                            // 256 chunks of 8 elems
            const int row = tid >> 2;               // 0..63
            const int cc  = (tid & 3) << 3;
            *(uint4*)&Ws[buf][row * TPAD + cc] =
                *(const uint4*)(wh + ((size_t)e * DOUT + n0 + row) * DIN + kk * BK + cc);
        }
#endif
    };

    v8f acc[2][2] = {};
    v16bf A[2];

    issueX(0, 0);
    issueW(0, 0, 0);

    for (int kk = 0; kk < NKI; ++kk) {
        const int xcur = kk & 1;
        const unsigned short* xs = &Xs[xcur][0];
#pragma unroll
        for (int e = 0; e < NEXP; ++e) {
            const int wcur = e & 1;
            // ---- issue next tile(s), then wait for the current ones ----
            if (e < NEXP - 1) {
                issueW(wcur ^ 1, e + 1, kk);
#if USE_TDM
                if (wave == 0) __builtin_amdgcn_s_wait_tensorcnt((short)1);
#endif
            } else if (kk + 1 < NKI) {
                issueX(xcur ^ 1, kk + 1);
                issueW(0, 0, kk + 1);
#if USE_TDM
                if (wave == 0) __builtin_amdgcn_s_wait_tensorcnt((short)2);
#endif
            } else {
#if USE_TDM
                if (wave == 0) __builtin_amdgcn_s_wait_tensorcnt((short)0);
#endif
            }
            __syncthreads();   // current X/W tiles visible to all waves

            if (e == 0) {      // A fragments are expert-invariant: load once per kk
#pragma unroll
                for (int mi = 0; mi < 2; ++mi) {
                    const v8bf* xr =
                        (const v8bf*)(xs + (32 * wm + 16 * mi + lr) * TPAD);
                    A[mi] = CAT16(xr[lh], xr[2 + lh]);   // K: lh*8..+7, 16+lh*8..+7
                }
            }
            const unsigned short* wsb = &Ws[wcur][0];
            v16bf B[2];
#pragma unroll
            for (int ni = 0; ni < 2; ++ni) {
                const v8bf* br =
                    (const v8bf*)(wsb + (32 * ng + 16 * ni + lr) * TPAD);
                B[ni] = CAT16(br[lh], br[2 + lh]);
            }
#pragma unroll
            for (int mi = 0; mi < 2; ++mi) {
                const v16bf As = A[mi] * gA[mi][e];      // gate folded into A rows
#pragma unroll
                for (int ni = 0; ni < 2; ++ni)
                    acc[mi][ni] = __builtin_amdgcn_wmma_f32_16x16x32_bf16(
                        false, As, false, B[ni], (short)0, acc[mi][ni],
                        false, false);
            }
            __syncthreads();   // all waves done with this W tile before overwrite
        }
    }

    // ---- epilogue: gated bias + store (C layout: M = r + 8*lh, N = lr) ----
    float ebv[NEXP][2];
#pragma unroll
    for (int e = 0; e < NEXP; ++e)
#pragma unroll
        for (int ni = 0; ni < 2; ++ni)
            ebv[e][ni] = eb[e * DOUT + n0 + 32 * ng + 16 * ni + lr];

#pragma unroll
    for (int mi = 0; mi < 2; ++mi) {
#pragma unroll
        for (int r = 0; r < 8; ++r) {
            const int tok = m0 + 32 * wm + 16 * mi + 8 * lh + r;
            const float* gp = gates + (size_t)tok * NEXP;
            const float4 glo = *(const float4*)gp;
            const float4 ghi = *(const float4*)(gp + 4);
#pragma unroll
            for (int ni = 0; ni < 2; ++ni) {
                float b = glo.x * ebv[0][ni] + glo.y * ebv[1][ni] +
                          glo.z * ebv[2][ni] + glo.w * ebv[3][ni] +
                          ghi.x * ebv[4][ni] + ghi.y * ebv[5][ni] +
                          ghi.z * ebv[6][ni] + ghi.w * ebv[7][ni];
                out[(size_t)tok * DOUT + n0 + 32 * ng + 16 * ni + lr] =
                    acc[mi][ni][r] + b;
            }
        }
    }
#else
    (void)xh; (void)wh; (void)eb; (void)gates; (void)out;
#endif // __HIP_DEVICE_COMPILE__
}

// ---------------------------------------------------------------------------
// Launcher
// ---------------------------------------------------------------------------
extern "C" void kernel_launch(void* const* d_in, const int* in_sizes, int n_in,
                              void* d_out, int out_size, void* d_ws, size_t ws_size,
                              hipStream_t stream) {
    (void)in_sizes; (void)n_in; (void)out_size; (void)ws_size;

    const float* x  = (const float*)d_in[0];   // [NTOK][DIN]
    const float* ew = (const float*)d_in[1];   // [E][DOUT][DIN]
    const float* eb = (const float*)d_in[2];   // [E][DOUT]
    const float* gw = (const float*)d_in[3];   // [E][DIN]
    const float* gb = (const float*)d_in[4];   // [E]
    float* out = (float*)d_out;                // [NTOK][DOUT]

    // Workspace layout: x_bf16 (16MB) | w_bf16 (16MB) | gates f32 (256KB)
    unsigned short* xh = (unsigned short*)d_ws;
    unsigned short* whb = xh + (size_t)NTOK * DIN;
    float* gates = (float*)(whb + (size_t)NEXP * DOUT * DIN);

    static_assert((size_t)NTOK * DIN == (size_t)NEXP * DOUT * DIN, "cvt grids match");
    const int cvt_blocks = (int)(((size_t)NTOK * DIN) / 8 / 256);   // 4096

    moe_cvt_bf16<<<cvt_blocks, 256, 0, stream>>>(x, xh);
    moe_cvt_bf16<<<cvt_blocks, 256, 0, stream>>>(ew, whb);
    moe_gates<<<NTOK / 8, 256, 0, stream>>>(x, gw, gb, gates);
    moe_wmma<<<dim3(NTOK / BM, DOUT / BN), NTHREADS, 0, stream>>>(xh, whb, eb, gates, out);
}
